// LSTM_63926293234137
// MI455X (gfx1250) — matrix operand order repaired and checked by reading, not compiled, not run on hardware
//
#include <hip/hip_runtime.h>
#include <hip/hip_bf16.h>

// ---------------- types ----------------
typedef __attribute__((ext_vector_type(16))) _Float16 v16h;
typedef __attribute__((ext_vector_type(8)))  _Float16 v8h;
typedef __attribute__((ext_vector_type(8)))  float    v8f;

#define NB 64      // batch
#define NT 256     // time steps
#define NE 512     // embed dim (layer-0 input)
#define NH 512     // hidden
#define NG 2048    // 4*H gate width

// ---------------- device helpers ----------------
__device__ __forceinline__ v8f wmma_f16(v16h a, v16h b, v8f c) {
    // D = A(16x32 f16) * B(32x16 f16) + C(16x16 f32)
    return __builtin_amdgcn_wmma_f32_16x16x32_f16(
        false, a, false, b, (short)0, c, false, false);
}

// A fragment (16x32, f16). Lane L: row m = base_m + (L&15), khalf = L>>4.
// p must point at rowptr + k + khalf*16 : 16 contiguous halves.
__device__ __forceinline__ v16h load_afrag(const _Float16* p) {
    v8h lo = *(const v8h*)p;
    v8h hi = *(const v8h*)(p + 8);
    return __builtin_shufflevector(lo, hi, 0,1,2,3,4,5,6,7,8,9,10,11,12,13,14,15);
}

// B fragment (32x16, f16) from row-major W[N][K] (W^T needed => column n = W row n).
// Lane L: col n = base_n + (L&15), khalf = L>>4.
// p must point at W + n*K + k + khalf*16 : 16 contiguous halves.
__device__ __forceinline__ v16h load_bfrag(const _Float16* p) {
    v8h lo = *(const v8h*)p;
    v8h hi = *(const v8h*)(p + 8);
    return __builtin_shufflevector(lo, hi, 0,1,2,3,4,5,6,7,8,9,10,11,12,13,14,15);
}

__device__ __forceinline__ float sigm(float x) { return 1.0f / (1.0f + __expf(-x)); }

// ---------------- prep kernels ----------------
__global__ void k_f2h(const float* __restrict__ s, _Float16* __restrict__ d, int n) {
    int i = blockIdx.x * blockDim.x + threadIdx.x;
    if (i < n) d[i] = (_Float16)s[i];
}

__global__ void k_bias_sum(const float* __restrict__ a, const float* __restrict__ b,
                           float* __restrict__ o, int n) {
    int i = blockIdx.x * blockDim.x + threadIdx.x;
    if (i < n) o[i] = a[i] + b[i];
}

__global__ void k_zero_u32(unsigned int* __restrict__ p, int n) {
    int i = blockIdx.x * blockDim.x + threadIdx.x;
    if (i < n) p[i] = 0u;
}

// ---------------- time-parallel input projection ----------------
// out[M, 2048] (f16) = A[M, 512] (f16) @ W[2048, 512]^T + bias[2048]
// One wave computes a 16x64 output tile. Block = 256 thr = 8 waves.
// grid.x = M/128 (8 m-tiles per block), grid.y = 2048/64 = 32.
__global__ __launch_bounds__(256)
void k_inproj(const _Float16* __restrict__ A, const _Float16* __restrict__ W,
              const float* __restrict__ bias, _Float16* __restrict__ out, int M) {
    const int tid   = threadIdx.x;
    const int wid   = tid >> 5;
    const int lane  = tid & 31;
    const int ln    = lane & 15;
    const int khalf = lane >> 4;

    const int mtile = blockIdx.x * 8 + wid;
    if (mtile * 16 >= M) return;              // wave-uniform
    const int nbase = blockIdx.y * 64;
    const int mrow  = mtile * 16 + ln;

    v8f acc[4] = {};
    const _Float16* arow = A + (size_t)mrow * 512;

    #pragma unroll 4
    for (int k = 0; k < 512; k += 32) {
        v16h af = load_afrag(arow + k + khalf * 16);
        #pragma unroll
        for (int s = 0; s < 4; ++s) {
            int n = nbase + s * 16 + ln;
            v16h bf = load_bfrag(W + (size_t)n * 512 + k + khalf * 16);
            acc[s] = wmma_f16(af, bf, acc[s]);
        }
    }
    #pragma unroll
    for (int s = 0; s < 4; ++s) {
        int n = nbase + s * 16 + ln;
        float bv = bias[n];
        #pragma unroll
        for (int j = 0; j < 8; ++j) {
            int m = mtile * 16 + khalf * 8 + j;
            out[(size_t)m * NG + n] = (_Float16)(acc[s][j] + bv);
        }
    }
}

// ---------------- fused recurrent step ----------------
struct RecArgs {
    const _Float16* xp[2];     // precomputed input projections (incl. bias), f16 [*, 2048]
    int             xpbase[2]; // xp row = xpbase + b * xpstride
    int             xpstride;
    const _Float16* hprev[2];  // [64,512] f16
    _Float16*       hout[2];   // [64,512] f16
    const float*    cprev[2];  // [64,512] f32
    float*          cout[2];   // [64,512] f32
    const _Float16* whh;       // [2048,512] f16
};

// grid = (32, 2): x = 16-col slice of H, y = direction. block = 128 thr = 4 waves.
// Wave mi handles batch rows 16*mi..16*mi+15 for ALL four gates of the same 16
// h-columns: 4 accumulators share one A-fragment (4x fewer A loads, 2.5 loads/WMMA).
__global__ __launch_bounds__(128)
void k_rec(RecArgs args) {
    __shared__ float g4[4][64][17];           // padded: kills LDS bank conflicts

    const int tid   = threadIdx.x;
    const int mi    = tid >> 5;               // wave id = m-tile
    const int lane  = tid & 31;
    const int ln    = lane & 15;
    const int khalf = lane >> 4;
    const int dir   = blockIdx.y;
    const int c0    = blockIdx.x * 16;        // column within H

    const float*    cp = args.cprev[dir];
    float*          co = args.cout[dir];
    _Float16*       ho = args.hout[dir];

    // Warm the pointwise-phase cell state while the GEMM runs:
    // threads 0..63 each prefetch one 64B row-chunk of c_prev (64 rows x 16 cols).
    if (tid < 64) __builtin_prefetch(&cp[tid * NH + c0], 0, 1);

    // ---- C init from precomputed input projection (f16 -> f32) ----
    const _Float16* xp = args.xp[dir];
    v8f acc[4];
    #pragma unroll
    for (int gi = 0; gi < 4; ++gi) {
        const int n_g = gi * NH + c0 + ln;
        #pragma unroll
        for (int j = 0; j < 8; ++j) {
            int b   = mi * 16 + khalf * 8 + j;    // batch row
            int row = args.xpbase[dir] + b * args.xpstride;
            acc[gi][j] = (float)xp[(size_t)row * NG + n_g];
        }
    }

    // ---- gates += h_prev @ Whh^T (A-fragment reused across 4 gates) ----
    const _Float16* ar = args.hprev[dir] + (size_t)(mi * 16 + ln) * 512;
    const _Float16* w0 = args.whh + (size_t)(0 * NH + c0 + ln) * 512;
    const _Float16* w1 = args.whh + (size_t)(1 * NH + c0 + ln) * 512;
    const _Float16* w2 = args.whh + (size_t)(2 * NH + c0 + ln) * 512;
    const _Float16* w3 = args.whh + (size_t)(3 * NH + c0 + ln) * 512;

    #pragma unroll 4
    for (int k = 0; k < 512; k += 32) {
        const int ko = k + khalf * 16;
        v16h af = load_afrag(ar + ko);
        acc[0] = wmma_f16(af, load_bfrag(w0 + ko), acc[0]);
        acc[1] = wmma_f16(af, load_bfrag(w1 + ko), acc[1]);
        acc[2] = wmma_f16(af, load_bfrag(w2 + ko), acc[2]);
        acc[3] = wmma_f16(af, load_bfrag(w3 + ko), acc[3]);
    }

    // ---- stash gate tiles to LDS ----
    #pragma unroll
    for (int gi = 0; gi < 4; ++gi) {
        #pragma unroll
        for (int j = 0; j < 8; ++j) {
            int b = mi * 16 + khalf * 8 + j;
            g4[gi][b][ln] = acc[gi][j];
        }
    }
    __syncthreads();

    // ---- LSTM pointwise: 64x16 = 1024 elements, 8 per thread ----
    #pragma unroll
    for (int e = tid; e < 1024; e += 128) {
        int b   = e >> 4;
        int cl  = e & 15;
        int col = c0 + cl;
        float iv = sigm(g4[0][b][cl]);
        float fv = sigm(g4[1][b][cl]);
        float gv = tanhf(g4[2][b][cl]);
        float ov = sigm(g4[3][b][cl]);
        float cn = fv * cp[b * NH + col] + iv * gv;
        co[b * NH + col] = cn;
        ho[b * NH + col] = (_Float16)(ov * tanhf(cn));
    }
}

// ---------------- final linear head ----------------
__global__ void k_final(const _Float16* __restrict__ hf, const _Float16* __restrict__ hr,
                        const float* __restrict__ wlin, const float* __restrict__ blin,
                        float* __restrict__ out) {
    int b = threadIdx.x;
    if (b >= NB) return;
    float acc = blin[0];
    for (int j = 0; j < NH; ++j) {
        acc += (float)hf[b * NH + j] * wlin[j];
        acc += (float)hr[b * NH + j] * wlin[NH + j];
    }
    out[b] = acc;
}

// ---------------- host side ----------------
extern "C" void kernel_launch(void* const* d_in, const int* in_sizes, int n_in,
                              void* d_out, int out_size, void* d_ws, size_t ws_size,
                              hipStream_t stream) {
    (void)in_sizes; (void)n_in; (void)out_size; (void)ws_size;

    const float* text  = (const float*)d_in[0];
    const float* Wih0  = (const float*)d_in[1];
    const float* Whh0  = (const float*)d_in[2];
    const float* bih0  = (const float*)d_in[3];
    const float* bhh0  = (const float*)d_in[4];
    const float* Wih1  = (const float*)d_in[5];
    const float* Whh1  = (const float*)d_in[6];
    const float* bih1  = (const float*)d_in[7];
    const float* bhh1  = (const float*)d_in[8];
    const float* Wlin  = (const float*)d_in[9];
    const float* blin  = (const float*)d_in[10];
    float* out = (float*)d_out;

    // ---- workspace carve-up ----
    char* base = (char*)d_ws;
    size_t off = 0;
    auto carve = [&](size_t bytes) -> char* {
        char* p = base + off;
        off = (off + bytes + 255) & ~(size_t)255;
        return p;
    };
    const size_t BH   = (size_t)NB * NH;                 // 32768
    const size_t MTOT = (size_t)NB * NT;                 // 16384 rows

    _Float16* xf16   = (_Float16*)carve(MTOT * NE * 2);          // text f16
    _Float16* wih0h  = (_Float16*)carve((size_t)NG * NE * 2);
    _Float16* whh0h  = (_Float16*)carve((size_t)NG * NH * 2);
    _Float16* wih1h  = (_Float16*)carve((size_t)NG * NH * 2);
    _Float16* whh1h  = (_Float16*)carve((size_t)NG * NH * 2);
    float*    bias0  = (float*)carve(NG * 4);
    float*    bias1  = (float*)carve(NG * 4);
    _Float16* xp0    = (_Float16*)carve(MTOT * NG * 2);          // 67 MB
    _Float16* xp1f   = (_Float16*)carve(MTOT * NG * 2);          // 67 MB
    _Float16* xp1r   = (_Float16*)carve(MTOT * NG * 2);          // 67 MB
    _Float16* hseq0  = (_Float16*)carve(2 * (size_t)NT * BH * 2);// layer-0 h sequence, both dirs
    _Float16* hping1 = (_Float16*)carve(2 * 2 * BH * 2);         // layer-1 h ping-pong
    float*    cping0 = (float*)carve(2 * 2 * BH * 4);
    float*    cping1 = (float*)carve(2 * 2 * BH * 4);
    _Float16* zh     = (_Float16*)carve(BH * 2);                 // zero h
    float*    zc     = (float*)carve(BH * 4);                    // zero c (contiguous with zh)

    // ---- prep: conversions, bias sums, zero state ----
    {
        int n = (int)(MTOT * NE);
        k_f2h<<<(n + 255) / 256, 256, 0, stream>>>(text, xf16, n);
        n = NG * NE;
        k_f2h<<<(n + 255) / 256, 256, 0, stream>>>(Wih0, wih0h, n);
        k_f2h<<<(n + 255) / 256, 256, 0, stream>>>(Whh0, whh0h, n);
        k_f2h<<<(n + 255) / 256, 256, 0, stream>>>(Wih1, wih1h, n);
        k_f2h<<<(n + 255) / 256, 256, 0, stream>>>(Whh1, whh1h, n);
        k_bias_sum<<<(NG + 255) / 256, 256, 0, stream>>>(bih0, bhh0, bias0, NG);
        k_bias_sum<<<(NG + 255) / 256, 256, 0, stream>>>(bih1, bhh1, bias1, NG);
        int zn = (int)((BH * 2 + 255 + BH * 4) / 4) + 64;        // zh (+pad) + zc, dwords
        k_zero_u32<<<(zn + 255) / 256, 256, 0, stream>>>((unsigned int*)zh, zn);
    }

    dim3 gemm_grid((unsigned)(MTOT / 128), NG / 64, 1);

    // ---- layer 0 input projection: XP0[b*T+t] = text @ Wih0^T + bias0 ----
    k_inproj<<<gemm_grid, 256, 0, stream>>>(xf16, wih0h, bias0, xp0, (int)MTOT);

    // ---- layer 0 recurrence, both directions fused ----
    for (int t = 0; t < NT; ++t) {
        RecArgs a;
        a.xp[0] = xp0;            a.xp[1] = xp0;
        a.xpbase[0] = t;          a.xpbase[1] = NT - 1 - t;
        a.xpstride  = NT;                                   // row = t_eff + b*T
        for (int d = 0; d < 2; ++d) {
            a.hprev[d] = (t == 0) ? zh : hseq0 + ((size_t)d * NT + (t - 1)) * BH;
            a.hout[d]  = hseq0 + ((size_t)d * NT + t) * BH;
            a.cprev[d] = (t == 0) ? zc : cping0 + ((size_t)d * 2 + ((t + 1) & 1)) * BH;
            a.cout[d]  = cping0 + ((size_t)d * 2 + (t & 1)) * BH;
        }
        a.whh = whh0h;
        k_rec<<<dim3(NH / 16, 2, 1), 128, 0, stream>>>(a);
    }

    // ---- layer 1 input projections (per direction) over layer-0 sequences ----
    k_inproj<<<gemm_grid, 256, 0, stream>>>(hseq0,                    wih1h, bias1, xp1f, (int)MTOT);
    k_inproj<<<gemm_grid, 256, 0, stream>>>(hseq0 + (size_t)NT * BH,  wih1h, bias1, xp1r, (int)MTOT);

    // ---- layer 1 recurrence, both directions fused ----
    for (int t = 0; t < NT; ++t) {
        RecArgs a;
        a.xp[0] = xp1f;           a.xp[1] = xp1r;
        a.xpbase[0] = t * NB;     a.xpbase[1] = t * NB;
        a.xpstride  = 1;                                    // row = t*64 + b
        for (int d = 0; d < 2; ++d) {
            a.hprev[d] = (t == 0) ? zh : hping1 + ((size_t)d * 2 + ((t + 1) & 1)) * BH;
            a.hout[d]  = hping1 + ((size_t)d * 2 + (t & 1)) * BH;
            a.cprev[d] = (t == 0) ? zc : cping1 + ((size_t)d * 2 + ((t + 1) & 1)) * BH;
            a.cout[d]  = cping1 + ((size_t)d * 2 + (t & 1)) * BH;
        }
        a.whh = whh1h;
        k_rec<<<dim3(NH / 16, 2, 1), 128, 0, stream>>>(a);
    }

    // ---- head: out[b] = concat(h_f, h_r) @ Wlin^T + blin ----
    const _Float16* h1f = hping1 + (size_t)(0 * 2 + ((NT - 1) & 1)) * BH;
    const _Float16* h1r = hping1 + (size_t)(1 * 2 + ((NT - 1) & 1)) * BH;
    k_final<<<1, 64, 0, stream>>>(h1f, h1r, Wlin, blin, out);
}
